// MultiAttentionLayer_24154896072806
// MI455X (gfx1250) — matrix operand order repaired and checked by reading, not compiled, not run on hardware
//
#include <hip/hip_runtime.h>
#include <hip/hip_bf16.h>

// ---------------------------------------------------------------------------
// MultiAttentionLayer for MI455X (gfx1250, wave32, WMMA bf16 16x16x32,
// async global->LDS staging)
// ---------------------------------------------------------------------------

typedef __attribute__((ext_vector_type(16))) __bf16 v16bf;
typedef __attribute__((ext_vector_type(8)))  __bf16 v8bf;
typedef __attribute__((ext_vector_type(8)))  float  v8f;

static constexpr int B  = 2;
static constexpr int S  = 2048;
static constexpr int E  = 1024;
static constexpr int H  = 16;
static constexpr int D  = 64;
static constexpr int HD = H * D;        // 1024
static constexpr int M  = B * S;        // 4096

// -------------------------- workspace layout (bytes) -----------------------
static constexpr size_t WS_QH   = 0;                 // bf16 [B*H][S][D]   8 MB
static constexpr size_t WS_KH   = 8ull  << 20;       // bf16 [B*H][S][D]   8 MB
static constexpr size_t WS_VHT  = 16ull << 20;       // bf16 [B*H][D][S]   8 MB
static constexpr size_t WS_WQT  = 24ull << 20;       // bf16 [N=1024][K=1024] 2 MB
static constexpr size_t WS_WKT  = 26ull << 20;
static constexpr size_t WS_WVT  = 28ull << 20;
static constexpr size_t WS_WOT  = 30ull << 20;
static constexpr size_t WS_CTX  = 32ull << 20;       // f32 [B*S][HD]     16 MB
static constexpr size_t WS_OPRE = 48ull << 20;       // f32 [B*S][E]      16 MB

// ------------------- async global->LDS helpers (gfx1250) -------------------
// Flat-aperture LDS pointers carry the LDS byte address in their low 32 bits
// (ISA 10.2), so a truncated generic pointer is a
// valid VDST lds-address operand for GLOBAL_LOAD_ASYNC_TO_LDS.
static __device__ inline void async_copy_b128(const void* gsrc, void* ldst) {
  const unsigned laddr = (unsigned)(uintptr_t)ldst;
  asm volatile("global_load_async_to_lds_b128 %0, %1, off"
               :: "v"(laddr), "v"(gsrc)
               : "memory");
}
static __device__ inline void async_wait_all() {
  asm volatile("s_wait_asynccnt 0x0" ::: "memory");
}

// ----------------------------- fragment helpers ----------------------------
// A fragment (16x32 bf16, MxK): lane holds row M=lane%16; K chunks
// [kbase, kbase+8) and [kbase+16, kbase+24), kbase = (lane<16 ? 0 : 8).
static __device__ inline v16bf load_frag_a_f32(const float* p) {
  v8f x = *reinterpret_cast<const v8f*>(p);
  v8f y = *reinterpret_cast<const v8f*>(p + 16);
  v16bf r;
#pragma unroll
  for (int i = 0; i < 8; ++i) { r[i] = (__bf16)x[i]; r[i + 8] = (__bf16)y[i]; }
  return r;
}

static __device__ inline v16bf load_frag_a_bf16(const __bf16* p) {
  v8bf x = *reinterpret_cast<const v8bf*>(p);
  v8bf y = *reinterpret_cast<const v8bf*>(p + 16);
  return __builtin_shufflevector(x, y, 0, 1, 2, 3, 4, 5, 6, 7,
                                        8, 9, 10, 11, 12, 13, 14, 15);
}

// B fragment (32x16 bf16, KxN): lane holds col N=lane%16; K chunks
// [kbase, kbase+8) and [kbase+16, kbase+24), kbase = (lane<16 ? 0 : 8). Source stored N-major (NxK).
static __device__ inline v16bf load_frag_b_bf16(const __bf16* p) {
  v8bf x = *reinterpret_cast<const v8bf*>(p);
  v8bf y = *reinterpret_cast<const v8bf*>(p + 16);
  return __builtin_shufflevector(x, y, 0, 1, 2, 3, 4, 5, 6, 7,
                                        8, 9, 10, 11, 12, 13, 14, 15);
}

static __device__ inline v8f wmma_bf16(v16bf a, v16bf b, v8f c) {
  return __builtin_amdgcn_wmma_f32_16x16x32_bf16(
      false, a, false, b, (short)0, c, false, false);
}

// ---------------------------------------------------------------------------
// Weight transpose+convert: in [K=1024][N=1024] f32 -> out [N][K] bf16
// ---------------------------------------------------------------------------
__global__ __launch_bounds__(256) void transpose_w_kernel(
    const float* __restrict__ in, __bf16* __restrict__ out) {
  const int idx = blockIdx.x * 256 + threadIdx.x;   // over 1024*1024
  const int k = idx >> 10;
  const int n = idx & 1023;
  out[(size_t)n * 1024 + k] = (__bf16)in[idx];
}

// ---------------------------------------------------------------------------
// Unified GEMM:  out = (A[Mx1024] * W^T[1024x1024] + bias) * scale
//   MODE 0: store bf16 head-major   qh/kh[b][h][s][d]
//   MODE 1: store bf16 transposed   vhT[b][h][d][s]
//   MODE 2: store f32 row-major     o_pre[m][n]
// Block tile 128x128, 8 waves, wave tile 32x64 (2x4 wmma frags), K step 32.
// For the Q projection scale=1/sqrt(D): folds the score scaling into qh so
// the QK^T kernel stores raw WMMA accumulators (no post-WMMA VALU hazard).
// ---------------------------------------------------------------------------
template <int MODE>
__global__ __launch_bounds__(256) void proj_gemm_kernel(
    const float* __restrict__ A, const __bf16* __restrict__ WT,
    const float* __restrict__ bias, void* __restrict__ outp, float scale) {
  const int tid  = threadIdx.x;
  const int wave = tid >> 5;
  const int lane = tid & 31;
  const int l16  = lane & 15;
  const int hi   = lane >> 4;

  const int mBlock = blockIdx.x * 128;
  const int nBlock = blockIdx.y * 128;
  const int wm = (wave & 3) * 32;
  const int wn = (wave >> 2) * 64;

  v8f acc[2][4] = {};

  for (int kb = 0; kb < 1024; kb += 32) {
    v16bf af[2];
#pragma unroll
    for (int mi = 0; mi < 2; ++mi) {
      const int row = mBlock + wm + mi * 16 + l16;
      af[mi] = load_frag_a_f32(A + (size_t)row * 1024 + kb + hi * 8);
    }
    v16bf bf[4];
#pragma unroll
    for (int ni = 0; ni < 4; ++ni) {
      const int col = nBlock + wn + ni * 16 + l16;
      bf[ni] = load_frag_b_bf16(WT + (size_t)col * 1024 + kb + hi * 8);
    }
#pragma unroll
    for (int mi = 0; mi < 2; ++mi)
#pragma unroll
      for (int ni = 0; ni < 4; ++ni)
        acc[mi][ni] = wmma_bf16(af[mi], bf[ni], acc[mi][ni]);
  }

#pragma unroll
  for (int mi = 0; mi < 2; ++mi) {
#pragma unroll
    for (int ni = 0; ni < 4; ++ni) {
      const int col = nBlock + wn + ni * 16 + l16;
      const float bv = bias[col];
#pragma unroll
      for (int r = 0; r < 8; ++r) {
        const int row = mBlock + wm + mi * 16 + r + hi * 8;
        const float v = (acc[mi][ni][r] + bv) * scale;
        if (MODE == 2) {
          ((float*)outp)[(size_t)row * 1024 + col] = v;
        } else {
          const int bb = row >> 11, ss = row & (S - 1);
          const int hh = col >> 6,  dd = col & 63;
          __bf16* o = (__bf16*)outp;
          if (MODE == 0)
            o[(((size_t)(bb * H + hh)) * S + ss) * D + dd] = (__bf16)v;
          else
            o[(((size_t)(bb * H + hh)) * D + dd) * S + ss] = (__bf16)v;
        }
      }
    }
  }
}

// ---------------------------------------------------------------------------
// scores[b,h,i,j] = qh[i,:] . kh[j,:]   (1/sqrt(D) pre-folded into qh)
// Causal tiles only if masked. Block tile 128(i) x 128(j), 8 waves of 32x64.
// ---------------------------------------------------------------------------
__global__ __launch_bounds__(256) void attn_scores_kernel(
    const __bf16* __restrict__ qh, const __bf16* __restrict__ kh,
    float* __restrict__ attn, const int* __restrict__ maskedp) {
  const int jb = blockIdx.x, ib = blockIdx.y, bh = blockIdx.z;
  if (*maskedp && jb > ib) return;   // uniform across block: EXEC stays full

  const int tid  = threadIdx.x;
  const int wave = tid >> 5;
  const int lane = tid & 31;
  const int l16  = lane & 15;
  const int hi   = lane >> 4;

  const __bf16* qb = qh + (size_t)bh * S * D;
  const __bf16* kp = kh + (size_t)bh * S * D;
  float* ab = attn + (size_t)bh * S * S;

  const int iBase = ib * 128 + (wave & 3) * 32;
  const int jBase = jb * 128 + (wave >> 2) * 64;

  v8f acc[2][4] = {};
#pragma unroll
  for (int kk = 0; kk < D; kk += 32) {
    v16bf af[2];
#pragma unroll
    for (int mi = 0; mi < 2; ++mi)
      af[mi] = load_frag_a_bf16(qb + (size_t)(iBase + mi * 16 + l16) * D + kk + hi * 8);
    v16bf bf[4];
#pragma unroll
    for (int ni = 0; ni < 4; ++ni)
      bf[ni] = load_frag_b_bf16(kp + (size_t)(jBase + ni * 16 + l16) * D + kk + hi * 8);
#pragma unroll
    for (int mi = 0; mi < 2; ++mi)
#pragma unroll
      for (int ni = 0; ni < 4; ++ni)
        acc[mi][ni] = wmma_bf16(af[mi], bf[ni], acc[mi][ni]);
  }

#pragma unroll
  for (int mi = 0; mi < 2; ++mi)
#pragma unroll
    for (int ni = 0; ni < 4; ++ni)
#pragma unroll
      for (int r = 0; r < 8; ++r) {
        const int i = iBase + mi * 16 + r + hi * 8;
        const int j = jBase + ni * 16 + l16;
        ab[(size_t)i * S + j] = acc[mi][ni][r];
      }
}

// ---------------------------------------------------------------------------
// Row softmax over j in [0, len); writes exact 0 for masked positions.
// Row staged memory->LDS with GLOBAL_LOAD_ASYNC_TO_LDS_B128 (ASYNCcnt).
// ---------------------------------------------------------------------------
__global__ __launch_bounds__(256) void softmax_kernel(
    float* __restrict__ attn, const int* __restrict__ maskedp) {
  __shared__ __align__(16) float buf[S];
  __shared__ float red[256];
  const int tid = threadIdx.x;
  const int i = blockIdx.x, bh = blockIdx.y;
  float* row = attn + (size_t)bh * S * S + (size_t)i * S;
  const int len = (*maskedp) ? (i + 1) : S;

  // async stage-in: 256 threads x 2 x 16B = 8 KB row
  async_copy_b128(row + tid * 4,        &buf[tid * 4]);
  async_copy_b128(row + 1024 + tid * 4, &buf[1024 + tid * 4]);
  async_wait_all();
  __syncthreads();

  float lmax = -3.0e38f;
  for (int j = tid; j < len; j += 256) lmax = fmaxf(lmax, buf[j]);
  red[tid] = lmax;
  __syncthreads();
  for (int off = 128; off > 0; off >>= 1) {
    if (tid < off) red[tid] = fmaxf(red[tid], red[tid + off]);
    __syncthreads();
  }
  const float mx = red[0];
  __syncthreads();

  float lsum = 0.f;
  for (int j = tid; j < len; j += 256) {
    const float e = __expf(buf[j] - mx);
    buf[j] = e;
    lsum += e;
  }
  red[tid] = lsum;
  __syncthreads();
  for (int off = 128; off > 0; off >>= 1) {
    if (tid < off) red[tid] += red[tid + off];
    __syncthreads();
  }
  const float inv = 1.0f / red[0];

  for (int j = tid; j < S; j += 256)
    row[j] = (j < len) ? buf[j] * inv : 0.f;
}

// ---------------------------------------------------------------------------
// ctx[b,s,h*D+d] = sum_j attn[b,h,s,j] * vh[b,j,h,d]  (vh stored as vhT[d][j])
// Block tile 128(i) x 64(d), 8 waves of 16x64. K-loop truncated at diagonal.
// ---------------------------------------------------------------------------
__global__ __launch_bounds__(256) void attn_v_kernel(
    const float* __restrict__ attn, const __bf16* __restrict__ vhT,
    float* __restrict__ ctx, const int* __restrict__ maskedp) {
  const int tid  = threadIdx.x;
  const int wave = tid >> 5;
  const int lane = tid & 31;
  const int l16  = lane & 15;
  const int hi   = lane >> 4;

  const int ib = blockIdx.x, bh = blockIdx.y;
  const int iBase = ib * 128 + wave * 16;
  const float*  ab = attn + (size_t)bh * S * S;
  const __bf16* vb = vhT + (size_t)bh * D * S;
  const int kEnd = (*maskedp) ? (ib * 128 + 128) : S;

  v8f acc[4] = {};
  for (int kb = 0; kb < kEnd; kb += 32) {
    const v16bf af =
        load_frag_a_f32(ab + (size_t)(iBase + l16) * S + kb + hi * 8);
    v16bf bf[4];
#pragma unroll
    for (int ni = 0; ni < 4; ++ni)
      bf[ni] = load_frag_b_bf16(vb + (size_t)(ni * 16 + l16) * S + kb + hi * 8);
#pragma unroll
    for (int ni = 0; ni < 4; ++ni)
      acc[ni] = wmma_bf16(af, bf[ni], acc[ni]);
  }

  const int bb = bh >> 4, hh = bh & 15;
#pragma unroll
  for (int ni = 0; ni < 4; ++ni)
#pragma unroll
    for (int r = 0; r < 8; ++r) {
      const int srow = iBase + r + hi * 8;
      const int dd = ni * 16 + l16;
      ctx[((size_t)(bb * S + srow)) * HD + hh * D + dd] = acc[ni][r];
    }
}

// ---------------------------------------------------------------------------
// outs = LayerNorm(o_pre + q), gamma=1, beta=0, eps=1e-5.
// Both source rows staged via async global->LDS copies.
// ---------------------------------------------------------------------------
__global__ __launch_bounds__(256) void layernorm_kernel(
    const float* __restrict__ o_pre, const float* __restrict__ q,
    float* __restrict__ outs) {
  __shared__ __align__(16) float bo_[E];
  __shared__ __align__(16) float bq_[E];
  __shared__ float red[256];
  const int tid = threadIdx.x;
  const int row = blockIdx.x;
  const float* op = o_pre + (size_t)row * E;
  const float* qp = q + (size_t)row * E;

  async_copy_b128(op + tid * 4, &bo_[tid * 4]);   // 256 x 16B = 4 KB
  async_copy_b128(qp + tid * 4, &bq_[tid * 4]);
  async_wait_all();
  __syncthreads();

  float lsum = 0.f;
  for (int e = tid; e < E; e += 256) {
    const float x = bo_[e] + bq_[e];
    bo_[e] = x;
    lsum += x;
  }
  red[tid] = lsum;
  __syncthreads();
  for (int off = 128; off > 0; off >>= 1) {
    if (tid < off) red[tid] += red[tid + off];
    __syncthreads();
  }
  const float mean = red[0] * (1.0f / E);
  __syncthreads();

  float lvar = 0.f;
  for (int e = tid; e < E; e += 256) {
    const float d2 = bo_[e] - mean;
    lvar += d2 * d2;
  }
  red[tid] = lvar;
  __syncthreads();
  for (int off = 128; off > 0; off >>= 1) {
    if (tid < off) red[tid] += red[tid + off];
    __syncthreads();
  }
  const float inv = rsqrtf(red[0] * (1.0f / E) + 1e-5f);

  for (int e = tid; e < E; e += 256)
    outs[(size_t)row * E + e] = (bo_[e] - mean) * inv;
}

// ---------------------------------------------------------------------------
extern "C" void kernel_launch(void* const* d_in, const int* in_sizes, int n_in,
                              void* d_out, int out_size, void* d_ws,
                              size_t ws_size, hipStream_t stream) {
  (void)in_sizes; (void)n_in; (void)out_size; (void)ws_size;

  const float* q  = (const float*)d_in[0];
  const float* k  = (const float*)d_in[1];
  const float* v  = (const float*)d_in[2];
  const float* Wq = (const float*)d_in[3];
  const float* bq = (const float*)d_in[4];
  const float* Wk = (const float*)d_in[5];
  const float* bk = (const float*)d_in[6];
  const float* Wv = (const float*)d_in[7];
  const float* bv = (const float*)d_in[8];
  const float* Wo = (const float*)d_in[9];
  const float* bo = (const float*)d_in[10];
  const int* masked = (const int*)d_in[11];

  float* outs = (float*)d_out;                       // [B,S,E]
  float* attn = outs + (size_t)B * S * E;            // [B,H,S,S]

  char* ws = (char*)d_ws;
  __bf16* qh   = (__bf16*)(ws + WS_QH);
  __bf16* kh   = (__bf16*)(ws + WS_KH);
  __bf16* vhT  = (__bf16*)(ws + WS_VHT);
  __bf16* WqT  = (__bf16*)(ws + WS_WQT);
  __bf16* WkT  = (__bf16*)(ws + WS_WKT);
  __bf16* WvT  = (__bf16*)(ws + WS_WVT);
  __bf16* WoT  = (__bf16*)(ws + WS_WOT);
  float* ctx   = (float*)(ws + WS_CTX);
  float* o_pre = (float*)(ws + WS_OPRE);

  // 1) weights -> bf16, N-major
  transpose_w_kernel<<<4096, 256, 0, stream>>>(Wq, WqT);
  transpose_w_kernel<<<4096, 256, 0, stream>>>(Wk, WkT);
  transpose_w_kernel<<<4096, 256, 0, stream>>>(Wv, WvT);
  transpose_w_kernel<<<4096, 256, 0, stream>>>(Wo, WoT);

  // 2) QKV projections (M=4096, N=1024, K=1024); Q absorbs 1/sqrt(D)
  const dim3 gg(M / 128, 1024 / 128);
  proj_gemm_kernel<0><<<gg, 256, 0, stream>>>(q, WqT, bq, (void*)qh, 0.125f);
  proj_gemm_kernel<0><<<gg, 256, 0, stream>>>(k, WkT, bk, (void*)kh, 1.0f);
  proj_gemm_kernel<1><<<gg, 256, 0, stream>>>(v, WvT, bv, (void*)vhT, 1.0f);

  // 3) scores -> attn region (raw), causal tiles only
  attn_scores_kernel<<<dim3(S / 128, S / 128, B * H), 256, 0, stream>>>(
      qh, kh, attn, masked);

  // 4) softmax in place (writes zeros above diagonal)
  softmax_kernel<<<dim3(S, B * H), 256, 0, stream>>>(attn, masked);

  // 5) attn @ V -> ctx
  attn_v_kernel<<<dim3(S / 128, B * H), 256, 0, stream>>>(attn, vhT, ctx,
                                                          masked);

  // 6) output projection
  proj_gemm_kernel<2><<<gg, 256, 0, stream>>>(ctx, WoT, bo, (void*)o_pre,
                                              1.0f);

  // 7) residual + layernorm
  layernorm_kernel<<<M, 256, 0, stream>>>(o_pre, q, outs);
}
